// SpikingJellyReservoirSNN_55465207660607
// MI455X (gfx1250) — matrix operand order repaired and checked by reading, not compiled, not run on hardware
//
#include <hip/hip_runtime.h>
#include <stdint.h>

// ---------------------------------------------------------------------------
// SpikingJelly reservoir SNN on gfx1250 (MI455X).
// Recurrent GEMM via v_wmma_f32_16x16x32_bf16 with hi/lo weight split
// (near-fp32 accuracy). Persistent kernel, decentralized flag barrier per
// timestep. Each wave computes TWO 16x16 M-tiles sharing one B-fragment,
// halving LDS bandwidth per WMMA (LDS was the per-step bottleneck).
// ---------------------------------------------------------------------------

typedef __bf16 bf16_t;
typedef __bf16 v16bf __attribute__((ext_vector_type(16)));
typedef float  v8f   __attribute__((ext_vector_type(8)));
typedef unsigned int v4u __attribute__((ext_vector_type(4)));

#define B_   128
#define C_   8
#define T_   3000
#define R_   1024
#define NWG  64
#define TPB  128                    // 4 wave32 waves, 2 M-tiles each
#define BPITCH (R_ + 8)             // LDS B row pitch (bf16 elems), 16B pad

// workspace layout (bytes)
#define WS_WHI 0
#define WS_WLO (2u*1024u*1024u)
#define WS_S0  (4u*1024u*1024u)
#define WS_S1  (4u*1024u*1024u + 256u*1024u)
#define WS_BAR (4u*1024u*1024u + 512u*1024u)

// ---------------------------------------------------------------------------
// Prep: split W_rec into bf16 hi + bf16 residual, zero spikes + flags.
// ---------------------------------------------------------------------------
__global__ void snn_prep(const float* __restrict__ Wrec,
                         bf16_t* __restrict__ whi, bf16_t* __restrict__ wlo,
                         bf16_t* __restrict__ s0, unsigned* __restrict__ bar)
{
    int i = blockIdx.x * blockDim.x + threadIdx.x;
    if (i < R_ * R_) {
        float w  = Wrec[i];
        bf16_t h = (bf16_t)w;
        whi[i]   = h;
        wlo[i]   = (bf16_t)(w - (float)h);
    }
    if (i < B_ * R_) s0[i] = (bf16_t)0.0f;
    if (i < NWG)     bar[i] = 0u;
}

// ---------------------------------------------------------------------------
static __device__ __forceinline__
void lif_update(v8f& acc, v8f& v, v8f& cnt, int mtile, int khalf, int r,
                bf16_t* __restrict__ snext)
{
#pragma unroll
    for (int j = 0; j < 8; ++j) {
        int m = mtile + j + 8 * khalf;                 // C/D layout row
        float vi = 0.5f * (v[j] + acc[j]);             // v += (I - v)/tau, tau=2
        float sp = (vi >= 1.0f) ? 1.0f : 0.0f;         // heaviside(v - 1)
        cnt[j] += sp;
        v[j] = vi - vi * sp;                           // hard reset to 0
        snext[m * R_ + r] = (bf16_t)sp;                // {0,1} exact in bf16
    }
}

static __device__ __forceinline__
v8f input_proj(const float* __restrict__ XS, int mtile, int khalf,
               float bias, float4 w0, float4 w1)
{
    v8f acc;
#pragma unroll
    for (int j = 0; j < 8; ++j) {
        int m = mtile + j + 8 * khalf;
        const float4* xr = (const float4*)&XS[m * C_];
        float4 a0 = xr[0], a1 = xr[1];
        acc[j] = bias
               + a0.x * w0.x + a0.y * w0.y + a0.z * w0.z + a0.w * w0.w
               + a1.x * w1.x + a1.y * w1.y + a1.z * w1.z + a1.w * w1.w;
    }
    return acc;
}

// ---------------------------------------------------------------------------
// Main persistent kernel.
//   64 WGs x 128 threads. WG owns 16 output columns; wave i owns rows
//   [i*32, i*32+32) as two 16x16 tiles sharing each B-fragment.
// ---------------------------------------------------------------------------
__launch_bounds__(TPB)
__global__ void snn_main(const float* __restrict__ x,
                         const float* __restrict__ Win,
                         const float* __restrict__ bin,
                         const float* __restrict__ brec,
                         const bf16_t* __restrict__ whi,
                         const bf16_t* __restrict__ wlo,
                         bf16_t* __restrict__ s0,
                         bf16_t* __restrict__ s1,
                         float* __restrict__ out,
                         unsigned* __restrict__ bar)
{
    extern __shared__ char smem[];
    bf16_t* BH = (bf16_t*)smem;                            // 16*BPITCH bf16
    bf16_t* BL = (bf16_t*)(smem + 16 * BPITCH * 2);        // 16*BPITCH bf16
    float*  XS = (float*) (smem + 2 * 16 * BPITCH * 2);    // 128*8 f32

    const int tid   = threadIdx.x;
    const int wg    = blockIdx.x;
    const int wave  = tid >> 5;
    const int lane  = tid & 31;
    const int nloc  = lane & 15;     // column within tile (C-layout: N = lane&15)
    const int khalf = lane >> 4;     // which K half-group this lane holds
    const int nbase = wg * 16;
    const int r     = nbase + nloc;  // this thread's output column
    const int m0    = wave * 32;     // first M tile
    const int m1    = m0 + 16;       // second M tile

    // ---- stage W_rec hi/lo rows for our 16 columns into LDS (once) ----
    for (int i = tid; i < 16 * (R_ / 8); i += TPB) {       // 16B chunks
        int n = i >> 7, ch = i & 127;
        *(v4u*)&BH[n * BPITCH + ch * 8] = *(const v4u*)&whi[(nbase + n) * R_ + ch * 8];
        *(v4u*)&BL[n * BPITCH + ch * 8] = *(const v4u*)&wlo[(nbase + n) * R_ + ch * 8];
    }

    // ---- per-thread constants: input-proj row of W_in + fused bias ----
    const float bias = bin[r] + brec[r];
    const float4 w0 = *(const float4*)&Win[r * C_ + 0];
    const float4 w1 = *(const float4*)&Win[r * C_ + 4];

    v8f v0, v1, c0, c1;
#pragma unroll
    for (int j = 0; j < 8; ++j) { v0[j] = v1[j] = c0[j] = c1[j] = 0.0f; }

    __syncthreads();

    const bf16_t* scur  = s0;
    bf16_t*       snext = s1;

    for (int t = 0; t < T_; ++t) {
        // ---- stage x_t [128 x 8] into LDS (x layout is [B,C,T]) ----
        for (int i = tid; i < B_ * C_; i += TPB)
            XS[i] = x[i * T_ + t];
        __syncthreads();

        // ---- seed accumulators with bias + input projection (f32 VALU) ----
        v8f acc0 = input_proj(XS, m0, khalf, bias, w0, w1);
        v8f acc1 = input_proj(XS, m1, khalf, bias, w0, w1);

        // ---- recurrent GEMM: acc += s * (W_hi + W_lo)^T via bf16 WMMA ----
        const bf16_t* ar0 = scur + (m0 + nloc) * R_;   // A lane row = lane&15
        const bf16_t* ar1 = scur + (m1 + nloc) * R_;
#pragma unroll 4
        for (int k = 0; k < R_; k += 32) {
            union { v16bf v; v4u q[2]; } a0, a1, bh, bl;
            // A 16x32 bf16: lanes 0-15 hold K 0..7 & 16..23; lanes 16-31: 8..15 & 24..31
            int ao = k + khalf * 8;
            a0.q[0] = *(const v4u*)&ar0[ao];
            a0.q[1] = *(const v4u*)&ar0[ao + 16];
            a1.q[0] = *(const v4u*)&ar1[ao];
            a1.q[1] = *(const v4u*)&ar1[ao + 16];
            // B 32x16 bf16: lane n (0-15) holds K 0..7 & 16..23; lanes 16-31: 8..15 & 24..31
            int bb = nloc * BPITCH + k + khalf * 8;
            bh.q[0] = *(const v4u*)&BH[bb];
            bh.q[1] = *(const v4u*)&BH[bb + 16];
            bl.q[0] = *(const v4u*)&BL[bb];
            bl.q[1] = *(const v4u*)&BL[bb + 16];
            // interleave independent acc chains; each B frag feeds 2 WMMAs
            acc0 = __builtin_amdgcn_wmma_f32_16x16x32_bf16(false, a0.v, false, bl.v,
                                                           (short)0, acc0, false, false);
            acc1 = __builtin_amdgcn_wmma_f32_16x16x32_bf16(false, a1.v, false, bl.v,
                                                           (short)0, acc1, false, false);
            acc0 = __builtin_amdgcn_wmma_f32_16x16x32_bf16(false, a0.v, false, bh.v,
                                                           (short)0, acc0, false, false);
            acc1 = __builtin_amdgcn_wmma_f32_16x16x32_bf16(false, a1.v, false, bh.v,
                                                           (short)0, acc1, false, false);
        }

        // ---- LIF update + spike emit (bf16, exact) ----
        lif_update(acc0, v0, c0, m0, khalf, r, snext);
        lif_update(acc1, v1, c1, m1, khalf, r, snext);

        // ---- swap spike buffers ----
        { const bf16_t* tmp = scur; scur = snext; snext = (bf16_t*)tmp; }

        // ---- decentralized grid barrier (release/acquire flags) ----
        __threadfence();            // each thread makes its spike stores visible
        __syncthreads();            // all threads in WG have fenced
        if (tid == 0)
            ((volatile unsigned*)bar)[wg] = (unsigned)(t + 1);   // release
        if (tid < NWG) {
            volatile unsigned* f = (volatile unsigned*)bar + tid;
            while (*f < (unsigned)(t + 1)) __builtin_amdgcn_s_sleep(2);
        }
        __threadfence();            // acquire
        __syncthreads();
    }

    // ---- emit spike counts [B x R] f32 ----
#pragma unroll
    for (int j = 0; j < 8; ++j) {
        out[(m0 + j + 8 * khalf) * R_ + r] = c0[j];
        out[(m1 + j + 8 * khalf) * R_ + r] = c1[j];
    }
}

// ---------------------------------------------------------------------------
extern "C" void kernel_launch(void* const* d_in, const int* in_sizes, int n_in,
                              void* d_out, int out_size, void* d_ws, size_t ws_size,
                              hipStream_t stream)
{
    const float* x    = (const float*)d_in[0];   // [128,8,3000]
    const float* Win  = (const float*)d_in[1];   // [1024,8]
    const float* bin  = (const float*)d_in[2];   // [1024]
    const float* Wrec = (const float*)d_in[3];   // [1024,1024]
    const float* brec = (const float*)d_in[4];   // [1024]
    float* out = (float*)d_out;                  // [128,1024]

    char* ws = (char*)d_ws;
    bf16_t*   whi = (bf16_t*)(ws + WS_WHI);
    bf16_t*   wlo = (bf16_t*)(ws + WS_WLO);
    bf16_t*   s0  = (bf16_t*)(ws + WS_S0);
    bf16_t*   s1  = (bf16_t*)(ws + WS_S1);
    unsigned* bar = (unsigned*)(ws + WS_BAR);

    snn_prep<<<(R_ * R_ + 255) / 256, 256, 0, stream>>>(Wrec, whi, wlo, s0, bar);

    size_t shmem = (size_t)(2 * 16 * BPITCH * 2) + (size_t)(B_ * C_ * 4);
    snn_main<<<NWG, TPB, shmem, stream>>>(x, Win, bin, brec, whi, wlo,
                                          s0, s1, out, bar);
}
